// QuantizedLinear_22823456211176
// MI455X (gfx1250) — matrix operand order repaired and checked by reading, not compiled, not run on hardware
//
#include <hip/hip_runtime.h>

// ---- types -----------------------------------------------------------------
typedef _Float16 h16;
typedef __attribute__((ext_vector_type(2)))  _Float16 v2h;
typedef __attribute__((ext_vector_type(8)))  _Float16 v8h;
typedef __attribute__((ext_vector_type(16))) _Float16 v16h;
typedef __attribute__((ext_vector_type(8)))  float    v8f;
typedef __attribute__((ext_vector_type(4)))  float    v4f;
typedef __attribute__((ext_vector_type(4)))  int      v4i;

union AFrag { v16h v; v8h h[2]; };

#define AS1 __attribute__((address_space(1)))
#define AS3 __attribute__((address_space(3)))

#if __has_builtin(__builtin_amdgcn_global_load_async_to_lds_b128) && \
    __has_builtin(__builtin_amdgcn_s_wait_asynccnt)
#define HAVE_ASYNC_LDS 1
#else
#define HAVE_ASYNC_LDS 0
#endif

// generic->AS3: low 32 bits of a generic LDS pointer are the LDS byte offset
__device__ __forceinline__ AS3 v4i* lds_cast(void* p) {
    return (AS3 v4i*)(unsigned)(unsigned long long)p;
}
__device__ __forceinline__ AS1 v4i* gbl_cast(const void* p) {
    return (AS1 v4i*)(unsigned long long)p;
}

// ---- tiling ----------------------------------------------------------------
#define BM  128   // rows of x per block
#define BN  128   // output features per block
#define BK  32    // K step (f16 WMMA K)
#define LDK 40    // padded LDS row stride in halves (80B: 16B-aligned, conflict-free)

// int4 nibble -> exact f16 (q-8), unscaled. 0x6400|q == (half)(1024+q) exactly.
__device__ __forceinline__ v2h dequant_pair(int v) {
    unsigned u = 0x64006400u | (unsigned)(v & 0xF) | ((unsigned)(v & 0xF0) << 12);
    v2h h = __builtin_bit_cast(v2h, u);
    return h + (v2h)((_Float16)-1032.0f);   // exact: (1024+q)-1032 = q-8
}

// ==== prep kernel 1: x fp32 -> f16 ==========================================
__global__ __launch_bounds__(256)
void prep_x_f16(const float* __restrict__ x, h16* __restrict__ xh, long n) {
    long i = ((long)blockIdx.x * 256 + threadIdx.x) * 8;
    if (i >= n) return;
    v4f a = *(const v4f*)&x[i];
    v4f b = *(const v4f*)&x[i + 4];
    v8h o;
#pragma unroll
    for (int j = 0; j < 4; ++j) { o[j] = (h16)a[j]; o[4 + j] = (h16)b[j]; }
    *(v8h*)&xh[i] = o;
}

// ==== prep kernel 2: packed int4 -> f16 (q-8), scale deferred to epilogue ===
__global__ __launch_bounds__(256)
void prep_w_f16(const int* __restrict__ wp, h16* __restrict__ wh, long nInts) {
    long i = ((long)blockIdx.x * 256 + threadIdx.x) * 4;
    if (i >= nInts) return;
    v4i v = *(const v4i*)&wp[i];
    v8h o;
#pragma unroll
    for (int e = 0; e < 4; ++e) {
        v2h h = dequant_pair(v[e]);
        o[2 * e + 0] = h[0];   // even k
        o[2 * e + 1] = h[1];   // odd  k
    }
    *(v8h*)&wh[2 * i] = o;
}

// ==== main GEMM: f16 WMMA + async global->LDS staging =======================
__global__ __launch_bounds__(256)
void gemm_f16_wmma(const h16*  __restrict__ xh,     // [N][IN]  f16
                   const h16*  __restrict__ wh,     // [OUT][IN] f16 (q-8)
                   const float* __restrict__ wscale,
                   const float* __restrict__ bias,
                   float*       __restrict__ out,   // [N][OUT]
                   int IN, int OUT)
{
    __shared__ __align__(16) h16 xs[2][BM * LDK];
    __shared__ __align__(16) h16 wsld[2][BN * LDK];

    const int t    = threadIdx.x;
    const int lane = t & 31;
    const int wv   = t >> 5;
    const int wM   = wv & 3;          // +32 rows
    const int wN   = wv >> 2;         // +64 cols

    const int mBase = blockIdx.y * BM;
    const int oBase = blockIdx.x * BN;

    // loader mapping: 128 rows x 4 segs of 8 halves (16B); 2 assignments/thread
    const int r0  = (t + 0)   >> 2, s0 = (t + 0)   & 3;
    const int r1  = (t + 256) >> 2, s1 = (t + 256) & 3;

    v8f acc[2][4];
#pragma unroll
    for (int mi = 0; mi < 2; ++mi)
#pragma unroll
        for (int ni = 0; ni < 4; ++ni) acc[mi][ni] = (v8f)(0.0f);

    const int laneM = lane & 15;
    const int laneH = lane >> 4;

    auto compute = [&](int b) {
        AFrag a[2];
#pragma unroll
        for (int mi = 0; mi < 2; ++mi) {
            // A 16x32: lanes0-15 K=0..7 & 16..23 ; lanes16-31 K=8..15 & 24..31
            const h16* p = &xs[b][(wM * 32 + mi * 16 + laneM) * LDK + laneH * 8];
            a[mi].h[0] = *(const v8h*)p;
            a[mi].h[1] = *(const v8h*)(p + 16);
        }
        AFrag bb[4];
#pragma unroll
        for (int ni = 0; ni < 4; ++ni) {
            // B 32x16: lane = N; lanes0-15 K=0..7 & 16..23 ; lanes16-31 K=8..15 & 24..31
            const h16* p = &wsld[b][(wN * 64 + ni * 16 + laneM) * LDK + laneH * 8];
            bb[ni].h[0] = *(const v8h*)p;
            bb[ni].h[1] = *(const v8h*)(p + 16);
        }
#pragma unroll
        for (int mi = 0; mi < 2; ++mi)
#pragma unroll
            for (int ni = 0; ni < 4; ++ni)
                acc[mi][ni] = __builtin_amdgcn_wmma_f32_16x16x32_f16(
                    false, a[mi].v, false, bb[ni].v, (short)0, acc[mi][ni],
                    false, false);
    };

    const int kTiles = IN / BK;

#if HAVE_ASYNC_LDS
    // DMA tile straight into LDS (ASYNCcnt path); no VGPR staging.
    auto asyncLoad = [&](int k0, int b) {
        __builtin_amdgcn_global_load_async_to_lds_b128(
            gbl_cast(&xh[(size_t)(mBase + r0) * IN + k0 + s0 * 8]),
            lds_cast(&xs[b][r0 * LDK + s0 * 8]), 0, 0);
        __builtin_amdgcn_global_load_async_to_lds_b128(
            gbl_cast(&xh[(size_t)(mBase + r1) * IN + k0 + s1 * 8]),
            lds_cast(&xs[b][r1 * LDK + s1 * 8]), 0, 0);
        __builtin_amdgcn_global_load_async_to_lds_b128(
            gbl_cast(&wh[(size_t)(oBase + r0) * IN + k0 + s0 * 8]),
            lds_cast(&wsld[b][r0 * LDK + s0 * 8]), 0, 0);
        __builtin_amdgcn_global_load_async_to_lds_b128(
            gbl_cast(&wh[(size_t)(oBase + r1) * IN + k0 + s1 * 8]),
            lds_cast(&wsld[b][r1 * LDK + s1 * 8]), 0, 0);
    };

    asyncLoad(0, 0);
    for (int kt = 0; kt < kTiles; ++kt) {
        const int b = kt & 1;
        __builtin_amdgcn_s_wait_asynccnt(0);  // this wave's tile landed in LDS
        __syncthreads();                      // all waves' tiles visible
        if (kt + 1 < kTiles) asyncLoad((kt + 1) * BK, b ^ 1);  // DMA overlaps WMMA
        compute(b);
    }
#else
    // fallback: register-staged double buffering
    v8h xv0, xv1, wv0, wv1;
    auto loadG = [&](int k0) {
        xv0 = *(const v8h*)&xh[(size_t)(mBase + r0) * IN + k0 + s0 * 8];
        xv1 = *(const v8h*)&xh[(size_t)(mBase + r1) * IN + k0 + s1 * 8];
        wv0 = *(const v8h*)&wh[(size_t)(oBase + r0) * IN + k0 + s0 * 8];
        wv1 = *(const v8h*)&wh[(size_t)(oBase + r1) * IN + k0 + s1 * 8];
    };
    auto storeL = [&](int b) {
        *(v8h*)&xs[b][r0 * LDK + s0 * 8]   = xv0;
        *(v8h*)&xs[b][r1 * LDK + s1 * 8]   = xv1;
        *(v8h*)&wsld[b][r0 * LDK + s0 * 8] = wv0;
        *(v8h*)&wsld[b][r1 * LDK + s1 * 8] = wv1;
    };
    loadG(0);
    storeL(0);
    __syncthreads();
    for (int kt = 0; kt < kTiles; ++kt) {
        const int b = kt & 1;
        if (kt + 1 < kTiles) loadG((kt + 1) * BK);
        compute(b);
        if (kt + 1 < kTiles) storeL(b ^ 1);
        __syncthreads();
    }
#endif

    // epilogue: per-column scale (deferred dequant) + bias
#pragma unroll
    for (int mi = 0; mi < 2; ++mi) {
#pragma unroll
        for (int ni = 0; ni < 4; ++ni) {
            const int row0 = mBase + wM * 32 + mi * 16 + laneH * 8;
            const int col  = oBase + wN * 64 + ni * 16 + laneM;
            const float s  = wscale[col];
            const float bv = bias[col];
#pragma unroll
            for (int j = 0; j < 8; ++j)
                out[(size_t)(row0 + j) * OUT + col] = fmaf(acc[mi][ni][j], s, bv);
        }
    }
}

// ==== fused fallback (if workspace too small): dequant in-loop ==============
__global__ __launch_bounds__(256)
void qlinear_fused_f16(const float* __restrict__ x,
                       const int*   __restrict__ wp,
                       const float* __restrict__ wscale,
                       const float* __restrict__ bias,
                       float*       __restrict__ out,
                       int IN, int OUT)
{
    __shared__ __align__(16) h16 xs[2][BM * LDK];
    __shared__ __align__(16) h16 wsld[2][BN * LDK];

    const int t    = threadIdx.x;
    const int lane = t & 31;
    const int wv   = t >> 5;
    const int wM   = wv & 3;
    const int wN   = wv >> 2;
    const int mBase = blockIdx.y * BM;
    const int oBase = blockIdx.x * BN;

    const int xr = t >> 3, xc = (t & 7) * 4;     // x: 4 float4 / thread
    const int wr = t >> 2, wc = (t & 3) * 4;     // w: 2 int4  / thread
    const int IN2 = IN >> 1;

    v4f xf[4];
    v4i wf[2];

    auto loadG = [&](int k0) {
#pragma unroll
        for (int r = 0; r < 4; ++r)
            xf[r] = *(const v4f*)&x[(size_t)(mBase + xr + 32 * r) * IN + k0 + xc];
#pragma unroll
        for (int r = 0; r < 2; ++r)
            wf[r] = *(const v4i*)&wp[(size_t)(oBase + wr + 64 * r) * IN2 + (k0 >> 1) + wc];
    };
    auto storeL = [&](int b) {
#pragma unroll
        for (int r = 0; r < 4; ++r) {
            h16* p = &xs[b][(xr + 32 * r) * LDK + xc];
#pragma unroll
            for (int j = 0; j < 4; ++j) p[j] = (h16)xf[r][j];
        }
#pragma unroll
        for (int r = 0; r < 2; ++r) {
            h16* p = &wsld[b][(wr + 64 * r) * LDK + wc * 2];
#pragma unroll
            for (int e = 0; e < 4; ++e)
                *(v2h*)(p + 2 * e) = dequant_pair(wf[r][e]);
        }
    };

    v8f acc[2][4];
#pragma unroll
    for (int mi = 0; mi < 2; ++mi)
#pragma unroll
        for (int ni = 0; ni < 4; ++ni) acc[mi][ni] = (v8f)(0.0f);

    const int laneM = lane & 15;
    const int laneH = lane >> 4;

    auto compute = [&](int b) {
        AFrag a[2];
#pragma unroll
        for (int mi = 0; mi < 2; ++mi) {
            const h16* p = &xs[b][(wM * 32 + mi * 16 + laneM) * LDK + laneH * 8];
            a[mi].h[0] = *(const v8h*)p;
            a[mi].h[1] = *(const v8h*)(p + 16);
        }
        AFrag bb[4];
#pragma unroll
        for (int ni = 0; ni < 4; ++ni) {
            const h16* p = &wsld[b][(wN * 64 + ni * 16 + laneM) * LDK + laneH * 8];
            bb[ni].h[0] = *(const v8h*)p;
            bb[ni].h[1] = *(const v8h*)(p + 16);
        }
#pragma unroll
        for (int mi = 0; mi < 2; ++mi)
#pragma unroll
            for (int ni = 0; ni < 4; ++ni)
                acc[mi][ni] = __builtin_amdgcn_wmma_f32_16x16x32_f16(
                    false, a[mi].v, false, bb[ni].v, (short)0, acc[mi][ni],
                    false, false);
    };

    const int kTiles = IN / BK;
    loadG(0);
    storeL(0);
    __syncthreads();
    for (int kt = 0; kt < kTiles; ++kt) {
        const int b = kt & 1;
        if (kt + 1 < kTiles) loadG((kt + 1) * BK);
        compute(b);
        if (kt + 1 < kTiles) storeL(b ^ 1);
        __syncthreads();
    }
#pragma unroll
    for (int mi = 0; mi < 2; ++mi) {
#pragma unroll
        for (int ni = 0; ni < 4; ++ni) {
            const int row0 = mBase + wM * 32 + mi * 16 + laneH * 8;
            const int col  = oBase + wN * 64 + ni * 16 + laneM;
            const float s  = wscale[col];
            const float bv = bias[col];
#pragma unroll
            for (int j = 0; j < 8; ++j)
                out[(size_t)(row0 + j) * OUT + col] = fmaf(acc[mi][ni][j], s, bv);
        }
    }
}

// ---- launcher --------------------------------------------------------------
extern "C" void kernel_launch(void* const* d_in, const int* in_sizes, int n_in,
                              void* d_out, int out_size, void* d_ws, size_t ws_size,
                              hipStream_t stream) {
    const float* x      = (const float*)d_in[0];
    const int*   wp     = (const int*)  d_in[1];
    const float* wscale = (const float*)d_in[2];
    const float* bias   = (const float*)d_in[3];
    float*       out    = (float*)d_out;

    const long OUT = in_sizes[2];
    const long IN  = 2L * in_sizes[1] / OUT;
    const long N   = in_sizes[0] / IN;

    const long xElems = N * IN;
    const long wInts  = OUT * IN / 2;
    const size_t need = (size_t)(xElems + OUT * IN) * sizeof(h16);

    dim3 grid((unsigned)(OUT / BN), (unsigned)(N / BM));

    if (ws_size >= need) {
        h16* xh = (h16*)d_ws;
        h16* wh = xh + xElems;
        prep_x_f16<<<(unsigned)(xElems / (256 * 8)), 256, 0, stream>>>(x, xh, xElems);
        prep_w_f16<<<(unsigned)(wInts  / (256 * 4)), 256, 0, stream>>>(wp, wh, wInts);
        gemm_f16_wmma<<<grid, 256, 0, stream>>>(xh, wh, wscale, bias, out,
                                                (int)IN, (int)OUT);
    } else {
        qlinear_fused_f16<<<grid, 256, 0, stream>>>(x, wp, wscale, bias, out,
                                                    (int)IN, (int)OUT);
    }
}
